// RHKH_ConvE_62998580297956
// MI455X (gfx1250) — matrix operand order repaired and checked by reading, not compiled, not run on hardware
//
#include <hip/hip_runtime.h>
#include <hip/hip_bf16.h>
#include <math.h>

typedef __bf16 bf16;
typedef __attribute__((ext_vector_type(16))) __bf16 v16bf;
typedef __attribute__((ext_vector_type(8)))  float  v8f;

#define N_ENT   50000
#define N_REL   500
#define DIM     200
#define DIMP    224            // DIM padded to multiple of 32 (K stride)
#define NPAD    256            // padded row count for small Bt weight buffers
#define N_EDGE  500000
#define BSZ     1024
#define NFILT   200
#define FLATSZ  39200
#define CONV_P  196            // 14*14 output pixels
#define CONV_M  (BSZ * CONV_P) // 200704 im2col rows
#define CONV_K  64             // 49 padded to 64
#define XROWS   (N_ENT + 48)   // x_bf oversized so score-GEMM Bt loads never OOB

union FragB { v16bf v; __bf16 e[16]; uint4 q[2]; };
union FragC { v8f  v; float  e[8]; };

// ---- WMMA GEMM: C[M x N] = A[M x K] * Bt[N x K]^T --------------------------
// All buffers are padded so EVERY load is an unguarded 16B global_load_b128:
//   - K multiple of 32, rows 16B aligned
//   - A has M rows (M multiple of 16), Bt has >= nQuads*64 rows (zero padding)
// Each wave computes a 16x64 strip: 1 A fragment feeds 4 WMMAs.
// MODE 0: tanh->bf16 | 1: none->bf16 | 2: +bias->bf16
// MODE 3: +bias,relu, scatter to fc layout [B x FLATSZ] (col=filter)
// MODE 4: +bias,relu->bf16 | 5: +bias,sigmoid->f32
template<int MODE>
__global__ void k_gemm(const bf16* __restrict__ A, const bf16* __restrict__ Bt,
                       const float* __restrict__ bias,
                       bf16* __restrict__ outB, float* __restrict__ outF,
                       int Ns /*out row stride*/, int Ng /*store/bias guard*/,
                       int K, int mTiles, int nQuads) {
  int lane = threadIdx.x & 31;
  int wid  = (int)((blockIdx.x * blockDim.x + threadIdx.x) >> 5);
  if (wid >= mTiles * nQuads) return;     // uniform per wave -> full EXEC at WMMA
  int mT = wid % mTiles;
  int nQ = wid / mTiles;

  // Contiguous k order used by this code (wave32; A and B match):
  //  A lane(half h, m=lane&15): elem i -> K = kb + 16*h + i
  //  B lane(halfK hk, n=lane&15): elem i -> K = kb + 16*hk + i
  const bf16* arow = A  + (size_t)(mT * 16 + (lane & 15)) * K + (size_t)((lane >> 4) * 16);
  const bf16* b0   = Bt + (size_t)(nQ * 64 + (lane & 15)) * K + (size_t)((lane >> 4) * 16);
  const bf16* b1   = b0 + (size_t)16 * K;
  const bf16* b2   = b0 + (size_t)32 * K;
  const bf16* b3   = b0 + (size_t)48 * K;

  v8f acc0 = {0.f,0.f,0.f,0.f,0.f,0.f,0.f,0.f};
  v8f acc1 = acc0, acc2 = acc0, acc3 = acc0;

  for (int kb = 0; kb < K; kb += 32) {
    FragB a, f0, f1, f2, f3;
    a.q[0]  = *(const uint4*)(arow + kb);
    a.q[1]  = *(const uint4*)(arow + kb + 8);
    f0.q[0] = *(const uint4*)(b0 + kb);  f0.q[1] = *(const uint4*)(b0 + kb + 8);
    f1.q[0] = *(const uint4*)(b1 + kb);  f1.q[1] = *(const uint4*)(b1 + kb + 8);
    f2.q[0] = *(const uint4*)(b2 + kb);  f2.q[1] = *(const uint4*)(b2 + kb + 8);
    f3.q[0] = *(const uint4*)(b3 + kb);  f3.q[1] = *(const uint4*)(b3 + kb + 8);
    // speculative prefetch of the next K chunks (global_prefetch_b8)
    __builtin_prefetch((const void*)(arow + kb + 128), 0, 1);
    __builtin_prefetch((const void*)(b0 + kb + 128), 0, 1);
    __builtin_prefetch((const void*)(b1 + kb + 128), 0, 1);
    __builtin_prefetch((const void*)(b2 + kb + 128), 0, 1);
    __builtin_prefetch((const void*)(b3 + kb + 128), 0, 1);
    acc0 = __builtin_amdgcn_wmma_f32_16x16x32_bf16(false, a.v, false, f0.v, (short)0, acc0, false, false);
    acc1 = __builtin_amdgcn_wmma_f32_16x16x32_bf16(false, a.v, false, f1.v, (short)0, acc1, false, false);
    acc2 = __builtin_amdgcn_wmma_f32_16x16x32_bf16(false, a.v, false, f2.v, (short)0, acc2, false, false);
    acc3 = __builtin_amdgcn_wmma_f32_16x16x32_bf16(false, a.v, false, f3.v, (short)0, acc3, false, false);
  }

  FragC c[4];
  c[0].v = acc0; c[1].v = acc1; c[2].v = acc2; c[3].v = acc3;
  int lh = lane >> 4, ln = lane & 15;
  for (int t = 0; t < 4; ++t) {
    int col = nQ * 64 + t * 16 + ln;
    if (col >= Ng) continue;                       // store-side guard only
    float bval = (MODE >= 2) ? bias[col] : 0.f;
    for (int r = 0; r < 8; ++r) {
      int row = mT * 16 + lh * 8 + r;
      float v = c[t].e[r] + bval;
      if (MODE == 0)      outB[(size_t)row * Ns + col] = (bf16)tanhf(c[t].e[r]);
      else if (MODE == 1) outB[(size_t)row * Ns + col] = (bf16)c[t].e[r];
      else if (MODE == 2) outB[(size_t)row * Ns + col] = (bf16)v;
      else if (MODE == 3) {
        v = v > 0.f ? v : 0.f;
        int b = row / CONV_P, p = row - b * CONV_P;
        outB[(size_t)b * FLATSZ + (size_t)col * CONV_P + p] = (bf16)v;
      }
      else if (MODE == 4) outB[(size_t)row * Ns + col] = (bf16)(v > 0.f ? v : 0.f);
      else                outF[(size_t)row * Ns + col] = 1.0f / (1.0f + __expf(-v));
    }
  }
}

// ---- graph message pass: one wave per edge, fp32 atomics into agg ----------
__global__ void k_edge(const float* __restrict__ x0, const float* __restrict__ r0,
                       const int* __restrict__ ei, const int* __restrict__ et,
                       float* __restrict__ agg, float* __restrict__ deg, int E) {
  int wid  = (int)((blockIdx.x * blockDim.x + threadIdx.x) >> 5);
  int lane = threadIdx.x & 31;
  if (wid >= E) return;
  int dst = ei[wid], src = ei[E + wid], t = et[wid];
  const float* xs = x0 + (size_t)src * DIM;
  const float* rt = r0 + (size_t)t * DIM;
  float* ad = agg + (size_t)dst * DIM;
  for (int d = lane; d < DIM; d += 32) atomicAdd(ad + d, xs[d] * rt[d]);
  if (lane == 0) atomicAdd(deg + dst, 1.0f);
}

// abar[n][0..223] = n<-- mean message (cols 200..223 zero)
__global__ void k_mean(const float* __restrict__ agg, const float* __restrict__ deg,
                       bf16* __restrict__ abar) {
  int i = blockIdx.x * blockDim.x + threadIdx.x;
  if (i >= N_ENT * DIMP) return;
  int n = i / DIMP, c = i - n * DIMP;
  float v = 0.f;
  if (c < DIM) {
    float d = deg[n]; d = d > 1.f ? d : 1.f;
    v = agg[n * DIM + c] / d;
  }
  abar[i] = (bf16)v;
}

// dst[Np x Kp] = padded bf16 transpose of src[R x C]: dst[n][k] = src[k][n]
__global__ void k_padT(const float* __restrict__ src, bf16* __restrict__ dst,
                       int R, int C, int Kp, int Np) {
  int i = blockIdx.x * blockDim.x + threadIdx.x;
  if (i >= Np * Kp) return;
  int n = i / Kp, k = i - n * Kp;
  dst[i] = (n < C && k < R) ? (bf16)src[(size_t)k * C + n] : (bf16)0.0f;
}

// rel0_bf [512 x 224] = padded bf16 copy of init_rel [500 x 200]
__global__ void k_pad_rel(const float* __restrict__ src, bf16* __restrict__ dst) {
  int i = blockIdx.x * blockDim.x + threadIdx.x;
  if (i >= 512 * DIMP) return;
  int r = i / DIMP, c = i - r * DIMP;
  dst[i] = (r < N_REL && c < DIM) ? (bf16)src[r * DIM + c] : (bf16)0.0f;
}

// convWT [256 x 64]: row f = filter, k zero-padded 49->64
__global__ void k_convw(const float* __restrict__ src, bf16* __restrict__ dst) {
  int i = blockIdx.x * blockDim.x + threadIdx.x;
  if (i >= NPAD * CONV_K) return;
  int f = i >> 6, k = i & 63;
  dst[i] = (f < NFILT && k < 49) ? (bf16)src[f * 49 + k] : (bf16)0.0f;
}

// bias2[n] = 5*shift_b[n] + sum_k shift_W[200+k][n]  (one-hot block collapses)
__global__ void k_bias2(const float* __restrict__ shift_W, const float* __restrict__ shift_b,
                        float* __restrict__ bias2) {
  int n = blockIdx.x * blockDim.x + threadIdx.x;
  if (n >= DIMP) return;
  float s = 0.f;
  if (n < DIM) {
    s = 5.0f * shift_b[n];
    for (int k = 0; k < 5; ++k) s += shift_W[(size_t)(DIM + k) * DIM + n];
  }
  bias2[n] = s;
}

__global__ void k_padvec(const float* __restrict__ src, float* __restrict__ dst) {
  int n = blockIdx.x * blockDim.x + threadIdx.x;
  if (n >= DIMP) return;
  dst[n] = (n < DIM) ? src[n] : 0.f;
}

// S[b][d] = sum_a x_add1[id2ent[sub[b]][a]][d]  (index N_ENT => zero row)
__global__ void k_gatherS(const bf16* __restrict__ x_bf, const int* __restrict__ sub,
                          const int* __restrict__ id2ent, bf16* __restrict__ S) {
  int i = blockIdx.x * blockDim.x + threadIdx.x;
  if (i >= BSZ * DIMP) return;
  int b = i / DIMP, d = i - b * DIMP;
  float acc = 0.f;
  if (d < DIM) {
    int s = sub[b];
    for (int a = 0; a < 5; ++a) {
      int e = id2ent[(size_t)s * 5 + a];
      if (e < N_ENT) acc += (float)x_bf[(size_t)e * DIMP + d];
    }
  }
  S[i] = (bf16)acc;
}

// im2col of interleaved [sub|rel] 20x20 image; K padded 49->64
__global__ void k_im2col(const bf16* __restrict__ sub_bf, const bf16* __restrict__ r_bf,
                         const int* __restrict__ rel, bf16* __restrict__ Aconv) {
  int i = blockIdx.x * blockDim.x + threadIdx.x;
  if (i >= CONV_M * CONV_K) return;
  int m = i >> 6, k = i & 63;
  bf16 val = (bf16)0.0f;
  if (k < 49) {
    int b  = m / CONV_P, p = m - b * CONV_P;
    int oh = p / 14, ow = p - oh * 14;
    int kh = k / 7,  kw = k - kh * 7;
    int f  = (oh + kh) * 20 + (ow + kw);   // flat pixel index = d*2 + c
    int d  = f >> 1;
    val = (f & 1) ? r_bf[(size_t)rel[b] * DIMP + d] : sub_bf[(size_t)b * DIMP + d];
  }
  Aconv[i] = val;
}

// ---- host orchestration -----------------------------------------------------
static inline int gemmBlocks(int mTiles, int nQuads) {
  long long waves = (long long)mTiles * nQuads;
  return (int)((waves + 7) / 8);  // 8 waves (256 threads) per block
}

extern "C" void kernel_launch(void* const* d_in, const int* in_sizes, int n_in,
                              void* d_out, int out_size, void* d_ws, size_t ws_size,
                              hipStream_t stream) {
  const float* init_embed = (const float*)d_in[0];
  const float* init_rel   = (const float*)d_in[1];
  const float* W_msg      = (const float*)d_in[2];
  const float* W_rel      = (const float*)d_in[3];
  const float* shift_W    = (const float*)d_in[4];
  const float* shift_b    = (const float*)d_in[5];
  const float* conv_W     = (const float*)d_in[6];
  const float* conv_b     = (const float*)d_in[7];
  const float* fc_W       = (const float*)d_in[8];
  const float* fc_b       = (const float*)d_in[9];
  const float* ent_bias   = (const float*)d_in[10];
  const int*   sub        = (const int*)d_in[11];
  const int*   rel        = (const int*)d_in[12];
  const int*   edge_index = (const int*)d_in[13];
  const int*   edge_type  = (const int*)d_in[14];
  const int*   id2ent     = (const int*)d_in[15];
  float* score = (float*)d_out;

  char* w = (char*)d_ws;
  auto carve = [&](size_t bytes) -> char* {
    char* p = w; w += (bytes + 255) & ~(size_t)255; return p;
  };
  float* agg     = (float*)carve((size_t)N_ENT * DIM * 4);
  float* deg     = (float*)carve((size_t)N_ENT * 4);
  bf16*  abar_bf = (bf16*) carve((size_t)N_ENT * DIMP * 2);
  bf16*  x_bf    = (bf16*) carve((size_t)XROWS * DIMP * 2);   // +48 zero rows for score Bt
  bf16*  WmsgT   = (bf16*) carve((size_t)NPAD * DIMP * 2);
  bf16*  WrelT   = (bf16*) carve((size_t)NPAD * DIMP * 2);
  bf16*  W1T     = (bf16*) carve((size_t)NPAD * DIMP * 2);
  float* bias2   = (float*)carve((size_t)DIMP * 4);
  float* fcb_pad = (float*)carve((size_t)DIMP * 4);
  bf16*  rel0_bf = (bf16*) carve((size_t)512 * DIMP * 2);
  bf16*  r_bf    = (bf16*) carve((size_t)512 * DIMP * 2);
  bf16*  S_bf    = (bf16*) carve((size_t)BSZ * DIMP * 2);
  bf16*  sub_bf  = (bf16*) carve((size_t)BSZ * DIMP * 2);
  bf16*  convWT  = (bf16*) carve((size_t)NPAD * CONV_K * 2);
  bf16*  Aconv   = (bf16*) carve((size_t)CONV_M * CONV_K * 2);
  bf16*  fcWT    = (bf16*) carve((size_t)NPAD * FLATSZ * 2);
  bf16*  Afc     = (bf16*) carve((size_t)BSZ * FLATSZ * 2);
  bf16*  h_bf    = (bf16*) carve((size_t)BSZ * DIMP * 2);

  const int TB = 256;
  // weight preps (padded; independent of graph phase)
  k_padT<<<(NPAD * DIMP + TB - 1) / TB, TB, 0, stream>>>(W_msg, WmsgT, DIM, DIM, DIMP, NPAD);
  k_padT<<<(NPAD * DIMP + TB - 1) / TB, TB, 0, stream>>>(W_rel, WrelT, DIM, DIM, DIMP, NPAD);
  k_padT<<<(NPAD * DIMP + TB - 1) / TB, TB, 0, stream>>>(shift_W, W1T, DIM, DIM, DIMP, NPAD);
  k_padT<<<(NPAD * FLATSZ + TB - 1) / TB, TB, 0, stream>>>(fc_W, fcWT, FLATSZ, DIM, FLATSZ, NPAD);
  k_convw<<<(NPAD * CONV_K + TB - 1) / TB, TB, 0, stream>>>(conv_W, convWT);
  k_bias2<<<1, TB, 0, stream>>>(shift_W, shift_b, bias2);
  k_padvec<<<1, TB, 0, stream>>>(fc_b, fcb_pad);
  k_pad_rel<<<(512 * DIMP + TB - 1) / TB, TB, 0, stream>>>(init_rel, rel0_bf);
  // zero x_bf oversize tail rows (rows N_ENT..XROWS-1)
  hipMemsetAsync(x_bf + (size_t)N_ENT * DIMP, 0, (size_t)(XROWS - N_ENT) * DIMP * 2, stream);

  // graph message pass
  hipMemsetAsync(agg, 0, (size_t)N_ENT * DIM * 4, stream);
  hipMemsetAsync(deg, 0, (size_t)N_ENT * 4, stream);
  k_edge<<<(N_EDGE * 32 + TB - 1) / TB, TB, 0, stream>>>(init_embed, init_rel,
                                                         edge_index, edge_type,
                                                         agg, deg, N_EDGE);
  k_mean<<<(N_ENT * DIMP + TB - 1) / TB, TB, 0, stream>>>(agg, deg, abar_bf);

  // x = tanh(mean @ W_msg)  [50000 x 224-padded]; pad cols naturally tanh(0)=0
  k_gemm<0><<<gemmBlocks(N_ENT / 16, 4), TB, 0, stream>>>(abar_bf, WmsgT, nullptr,
      x_bf, nullptr, DIMP, DIMP, DIMP, N_ENT / 16, 4);
  // r = init_rel @ W_rel  [512 x 224-padded]
  k_gemm<1><<<gemmBlocks(512 / 16, 4), TB, 0, stream>>>(rel0_bf, WrelT, nullptr,
      r_bf, nullptr, DIMP, DIMP, DIMP, 512 / 16, 4);
  // sub_emb = S @ W1 + bias2  [1024 x 224-padded]
  k_gatherS<<<(BSZ * DIMP + TB - 1) / TB, TB, 0, stream>>>(x_bf, sub, id2ent, S_bf);
  k_gemm<2><<<gemmBlocks(BSZ / 16, 4), TB, 0, stream>>>(S_bf, W1T, bias2,
      sub_bf, nullptr, DIMP, DIMP, DIMP, BSZ / 16, 4);
  // ConvE conv as im2col GEMM -> relu -> scatter to fc layout [1024 x 39200]
  k_im2col<<<(CONV_M * CONV_K + TB - 1) / TB, TB, 0, stream>>>(sub_bf, r_bf, rel, Aconv);
  k_gemm<3><<<gemmBlocks(CONV_M / 16, 4), TB, 0, stream>>>(Aconv, convWT, conv_b,
      Afc, nullptr, 0, NFILT, CONV_K, CONV_M / 16, 4);
  // h = relu(feat @ fc_W + fc_b)  [1024 x 224-padded]
  k_gemm<4><<<gemmBlocks(BSZ / 16, 4), TB, 0, stream>>>(Afc, fcWT, fcb_pad,
      h_bf, nullptr, DIMP, DIMP, FLATSZ, BSZ / 16, 4);
  // score = sigmoid(h @ x.T + ent_bias)  [1024 x 50000] f32 -> d_out
  int scoreQuads = (N_ENT + 63) / 64;  // 782; Bt rows up to 50047 valid (zeroed)
  k_gemm<5><<<gemmBlocks(BSZ / 16, scoreQuads), TB, 0, stream>>>(h_bf, x_bf, ent_bias,
      nullptr, score, N_ENT, N_ENT, DIMP, BSZ / 16, scoreQuads);
}
